// ETCLayer_19224273617225
// MI455X (gfx1250) — matrix operand order repaired and checked by reading, not compiled, not run on hardware
//
#include <hip/hip_runtime.h>
#include <hip/hip_bf16.h>
#include <math.h>

// ---------------------------------------------------------------------------
// MI455X (gfx1250) implementation of the ETC-style layer.
// Dense math on v_wmma_f32_16x16x32_f16 (wave32 WMMA); GEMM staging uses the
// CDNA5 async global->LDS path (ASYNCcnt) when the toolchain exposes it.
// ---------------------------------------------------------------------------

typedef __attribute__((ext_vector_type(8)))  _Float16 v8h;
typedef __attribute__((ext_vector_type(16))) _Float16 v16h;
typedef __attribute__((ext_vector_type(8)))  float    v8f;
typedef __attribute__((ext_vector_type(4)))  int      v4i;

constexpr int Bb   = 2;
constexpr int Ss   = 2048;
constexpr int Dm   = 1024;
constexpr int NH   = 16;
constexpr int DHd  = 64;
constexpr int DFFn = 4096;
constexpr int Mrows = Bb * Ss;       // 4096 token rows

#define LDS_AS __attribute__((address_space(3)))
#define GLB_AS __attribute__((address_space(1)))

#if __has_builtin(__builtin_amdgcn_global_load_async_to_lds_b128)
#define HAS_ASYNC_LDS 1
#else
#define HAS_ASYNC_LDS 0
#endif

typedef GLB_AS v4i* gv4i_p;
typedef LDS_AS v4i* lv4i_p;

static __device__ __forceinline__ void wait_async_zero() {
#if HAS_ASYNC_LDS
#if __has_builtin(__builtin_amdgcn_s_wait_asynccnt)
  __builtin_amdgcn_s_wait_asynccnt(0);
#else
  asm volatile("s_wait_asynccnt 0x0" ::: "memory");
#endif
#endif
}

static __device__ __forceinline__ void copy16_g2l(const _Float16* g, _Float16* l) {
#if HAS_ASYNC_LDS
  __builtin_amdgcn_global_load_async_to_lds_b128(
      (gv4i_p)(uintptr_t)g, (lv4i_p)(uintptr_t)l, 0, 0);
#else
  *(v8h*)l = *(const v8h*)g;
#endif
}

static __device__ __forceinline__ v16h cat8(v8h lo, v8h hi) {
  return __builtin_shufflevector(lo, hi, 0,1,2,3,4,5,6,7,8,9,10,11,12,13,14,15);
}
static __device__ __forceinline__ v8f wmma32(v16h a, v16h b, v8f c) {
  // D = A(16x32) x B(32x16) + C, fp32 accumulate
  return __builtin_amdgcn_wmma_f32_16x16x32_f16(false, a, false, b, (short)0, c, false, false);
}

// ---------------------------------------------------------------------------
// fp32 -> f16 cast
// ---------------------------------------------------------------------------
__global__ void cast_f2h(const float* __restrict__ in, _Float16* __restrict__ out, int n) {
  int i = blockIdx.x * 256 + threadIdx.x;
  if (i < n) out[i] = (_Float16)in[i];
}

// ---------------------------------------------------------------------------
// WMMA GEMM: C[M,N] = A[M,K] * W[N,K]^T + bias[N]   (A,W f16 row-major)
// Block: 128 threads (4 waves) -> 128x128 C tile; wave -> 64x64 (4x4 WMMA tiles)
// K-tile 64, double-buffered LDS, async global->LDS staging, 1 barrier/K-tile.
// ---------------------------------------------------------------------------
template<bool RELU, bool OUTH>
__global__ __launch_bounds__(128) void gemm_kernel(
    const _Float16* __restrict__ A, const _Float16* __restrict__ W,
    const float* __restrict__ bias,
    float* __restrict__ outF, _Float16* __restrict__ outH,
    int M, int N, int K)
{
  __shared__ _Float16 As[2][128][72];   // 64 K-cols + pad (144B row stride)
  __shared__ _Float16 Ws[2][128][72];

  const int tid  = threadIdx.x;
  const int lane = tid & 31;
  const int wave = tid >> 5;
  const int mBase = blockIdx.y * 128;
  const int nBase = blockIdx.x * 128;
  const int m0 = (wave >> 1) * 64;     // wave grid 2(M) x 2(N)
  const int n0 = (wave & 1) * 64;

  const int lr = lane & 15;
  const int lh = lane >> 4;
  const int r0 = tid >> 3;             // staging: 0..15
  const int c0 = (tid & 7) * 8;        // staging col: 0,8,...,56

  v8f acc[4][4];
  for (int i = 0; i < 4; ++i)
    for (int j = 0; j < 4; ++j)
      acc[i][j] = {};

  auto stage = [&](int buf, int k0) {
    const _Float16* Ab = A + (size_t)(mBase + r0) * K + k0 + c0;
    const _Float16* Wb = W + (size_t)(nBase + r0) * K + k0 + c0;
    for (int i = 0; i < 8; ++i) {
      int row = r0 + i * 16;
      copy16_g2l(Ab + (size_t)(i * 16) * K, &As[buf][row][c0]);
      copy16_g2l(Wb + (size_t)(i * 16) * K, &Ws[buf][row][c0]);
    }
  };

  const int nt = K >> 6;               // K / 64
  stage(0, 0);

  for (int t = 0; t < nt; ++t) {
    wait_async_zero();                 // our async loads for tile t complete
    __syncthreads();                   // everyone's loads visible; prev compute done
    if (t + 1 < nt) stage((t + 1) & 1, (t + 1) << 6);

    const int buf = t & 1;
    for (int ks = 0; ks < 64; ks += 32) {
      v16h af[4], bf[4];
      for (int i = 0; i < 4; ++i) {
        int row = m0 + i * 16 + lr;    // A layout: lane=row; K contiguous 16/half
        af[i] = cat8(*(const v8h*)&As[buf][row][ks + lh * 16],
                     *(const v8h*)&As[buf][row][ks + lh * 16 + 8]);
      }
      for (int j = 0; j < 4; ++j) {
        int row = n0 + j * 16 + lr;    // B layout: lane=col; K contiguous 16/half
        bf[j] = cat8(*(const v8h*)&Ws[buf][row][ks + lh * 16],
                     *(const v8h*)&Ws[buf][row][ks + lh * 16 + 8]);
      }
      for (int i = 0; i < 4; ++i)
        for (int j = 0; j < 4; ++j)
          acc[i][j] = wmma32(af[i], bf[j], acc[i][j]);
    }
  }

  // Epilogue: C layout lane -> n = lane&15, m = r + 8*(lane>>4)
  for (int j = 0; j < 4; ++j) {
    int n = nBase + n0 + j * 16 + lr;
    float bv = bias[n];
    for (int i = 0; i < 4; ++i) {
      for (int r = 0; r < 8; ++r) {
        int m = mBase + m0 + i * 16 + 8 * lh + r;
        float v = acc[i][j][r] + bv;
        if (RELU) v = fmaxf(v, 0.0f);
        if (OUTH) outH[(size_t)m * N + n] = (_Float16)v;
        else      outF[(size_t)m * N + n] = v;
      }
    }
  }
}

// ---------------------------------------------------------------------------
// Transpose V: (b,s,h*64+dh) -> Vt[(b*16+h)*64+dh][s]   (for contiguous B frags)
// ---------------------------------------------------------------------------
__global__ __launch_bounds__(256) void transpose_v(
    const _Float16* __restrict__ v, _Float16* __restrict__ vt)
{
  __shared__ _Float16 tile[32][33];
  const int bh = blockIdx.z;
  const int b = bh >> 4, h = bh & 15;
  const int s0 = blockIdx.x * 32, d0 = blockIdx.y * 32;
  const int tx = threadIdx.x, ty = threadIdx.y;     // 32 x 8
  for (int i = 0; i < 4; ++i) {
    int s = s0 + ty + i * 8;
    tile[ty + i * 8][tx] = v[((size_t)(b * Ss + s)) * Dm + h * 64 + d0 + tx];
  }
  __syncthreads();
  for (int i = 0; i < 4; ++i) {
    int d = d0 + ty + i * 8;
    vt[((size_t)(bh * 64 + d)) * Ss + s0 + tx] = tile[tx][ty + i * 8];
  }
}

// ---------------------------------------------------------------------------
// Windowed relative attention, one wave per 16-query block.
// scores = Q x K^T (5 key tiles of 16 covering window), rel bias via WMMA+shfl,
// softmax in VALU, attn bounced through LDS, out = attn x V via WMMA.
// ---------------------------------------------------------------------------
__global__ __launch_bounds__(256) void attn_kernel(
    const _Float16* __restrict__ Q, const _Float16* __restrict__ Kmat,
    const _Float16* __restrict__ Vt, const _Float16* __restrict__ Rel,
    _Float16* __restrict__ Out)
{
  __shared__ _Float16 attnLds[8][16][104];   // per wave: 16 x 96 (+8 pad halves)

  const int lane = threadIdx.x & 31;
  const int wave = threadIdx.x >> 5;
  const int id   = blockIdx.x * 8 + wave;    // 4096 qblocks total
  const int qb   = id & 127;                 // S/16 = 128
  const int bh   = id >> 7;                  // 0..31
  const int h    = bh & 15, b = bh >> 4;
  const int qbase = qb * 16;
  const int kbase = qbase - 32;              // 96 padded keys: kbase .. kbase+95

  const int lr = lane & 15;
  const int lh = lane >> 4;

  // zero attn pad columns 80..95 (keeps 0 * V finite in final WMMA)
  for (int i = lane; i < 256; i += 32)
    attnLds[wave][i >> 4][80 + (i & 15)] = (_Float16)0.0f;

  // ---- Q fragments (A layout), dh chunks 0..31 and 32..63
  const _Float16* qrow = Q + ((size_t)(b * Ss + qbase + lr)) * Dm + h * 64;
  v16h qf[2];
  for (int c = 0; c < 2; ++c)
    qf[c] = cat8(*(const v8h*)(qrow + c * 32 + lh * 16),
                 *(const v8h*)(qrow + c * 32 + lh * 16 + 8));

  // ---- scores: 5 key tiles of 16
  v8f sc[5];
  for (int t = 0; t < 5; ++t) {
    int j = kbase + t * 16 + lr;
    j = min(max(j, 0), Ss - 1);
    const _Float16* krow = Kmat + ((size_t)(b * Ss + j)) * Dm + h * 64;
    v8f a = {};
    for (int c = 0; c < 2; ++c) {
      v16h bf = cat8(*(const v8h*)(krow + c * 32 + lh * 16),
                     *(const v8h*)(krow + c * 32 + lh * 16 + 8));
      a = wmma32(qf[c], bf, a);
    }
    sc[t] = a;
  }

  // ---- relative bias: relscores(16x11) = Q x rel_emb^T
  v8f rs = {};
  {
    const _Float16* rrow = Rel + min(lr, 10) * 64;
    for (int c = 0; c < 2; ++c) {
      v16h bf = cat8(*(const v8h*)(rrow + c * 32 + lh * 16),
                     *(const v8h*)(rrow + c * 32 + lh * 16 + 8));
      rs = wmma32(qf[c], bf, rs);
    }
  }

  // ---- combine + mask + softmax; write probs to LDS in attn-col layout
  const float scale = 0.125f;                // 1/sqrt(64)
  for (int r = 0; r < 8; ++r) {
    int m = r + 8 * lh;                      // query row of this lane for vgpr r
    float rowmax = -3.0e38f;
    for (int t = 0; t < 5; ++t) {
      int off = t * 16 + lr - m - 32;        // j - s
      int rid = min(max(off, -5), 5) + 5;    // clip to [-K,K]+K
      float rb = __shfl(rs[r], rid + (lh << 4), 32);
      float s  = (sc[t][r] + rb) * scale;
      int j = qbase + m + off;
      bool valid = (off >= -25) & (off <= 25) & (j >= 0) & (j < Ss);
      s = valid ? s : -1.0e30f;
      sc[t][r] = s;
      rowmax = fmaxf(rowmax, s);
    }
    for (int d = 1; d < 16; d <<= 1)
      rowmax = fmaxf(rowmax, __shfl_xor(rowmax, d, 32));
    float se = 0.0f;
    for (int t = 0; t < 5; ++t) {
      float e = __expf(sc[t][r] - rowmax);
      sc[t][r] = e;
      se += e;
    }
    for (int d = 1; d < 16; d <<= 1)
      se += __shfl_xor(se, d, 32);
    float inv = 1.0f / se;
    for (int t = 0; t < 5; ++t)
      attnLds[wave][m][t * 16 + lr] = (_Float16)(sc[t][r] * inv);
  }

  __syncthreads();  // uniform; guarantee LDS visibility before fragment reads

  // ---- out(16x64) = attn(16x96) x V(96x64), 3 K-chunks of 32
  v8f oacc[4];
  for (int nt = 0; nt < 4; ++nt) oacc[nt] = {};
  for (int c = 0; c < 3; ++c) {
    v16h af = cat8(*(const v8h*)&attnLds[wave][lr][c * 32 + lh * 16],
                   *(const v8h*)&attnLds[wave][lr][c * 32 + lh * 16 + 8]);
    int js = kbase + c * 32 + lh * 16;       // 16-aligned; clamp keeps loads in-bounds
    js = min(max(js, 0), Ss - 16);
    for (int nt = 0; nt < 4; ++nt) {
      const _Float16* vrow = Vt + ((size_t)(bh * 64 + nt * 16 + lr)) * Ss + js;
      v16h bf = cat8(*(const v8h*)vrow, *(const v8h*)(vrow + 8));
      oacc[nt] = wmma32(af, bf, oacc[nt]);
    }
  }

  for (int nt = 0; nt < 4; ++nt)
    for (int r = 0; r < 8; ++r) {
      int m = r + 8 * lh;
      Out[((size_t)(b * Ss + qbase + m)) * Dm + h * 64 + nt * 16 + lr] = (_Float16)oacc[nt][r];
    }
}

// ---------------------------------------------------------------------------
// Residual + LayerNorm over 1024 cols, one block (256 threads) per row.
// ---------------------------------------------------------------------------
__global__ __launch_bounds__(256) void ln_kernel(
    const float* __restrict__ a, const float* __restrict__ b,
    const float* __restrict__ g, const float* __restrict__ beta,
    float* __restrict__ outF, _Float16* __restrict__ outH)
{
  __shared__ float redS[256];
  __shared__ float redQ[256];
  const int row = blockIdx.x, tid = threadIdx.x;
  const size_t base = (size_t)row * Dm + tid * 4;
  float4 av = *(const float4*)(a + base);
  float4 bv = *(const float4*)(b + base);
  float x[4] = { av.x + bv.x, av.y + bv.y, av.z + bv.z, av.w + bv.w };
  float s = x[0] + x[1] + x[2] + x[3];
  float q = x[0]*x[0] + x[1]*x[1] + x[2]*x[2] + x[3]*x[3];
  redS[tid] = s; redQ[tid] = q;
  __syncthreads();
  for (int off = 128; off > 0; off >>= 1) {
    if (tid < off) { redS[tid] += redS[tid + off]; redQ[tid] += redQ[tid + off]; }
    __syncthreads();
  }
  const float mean = redS[0] * (1.0f / Dm);
  const float var  = redQ[0] * (1.0f / Dm) - mean * mean;
  const float inv  = rsqrtf(var + 1e-5f);
  for (int e = 0; e < 4; ++e) {
    int col = tid * 4 + e;
    float y = (x[e] - mean) * inv * g[col] + beta[col];
    if (outF) outF[base + e] = y;
    if (outH) outH[base + e] = (_Float16)y;
  }
}

// ---------------------------------------------------------------------------
// Host orchestration
// ---------------------------------------------------------------------------
extern "C" void kernel_launch(void* const* d_in, const int* in_sizes, int n_in,
                              void* d_out, int out_size, void* d_ws, size_t ws_size,
                              hipStream_t stream)
{
  const float* src = (const float*)d_in[0];
  const float* wq  = (const float*)d_in[1];
  const float* bq  = (const float*)d_in[2];
  const float* wk  = (const float*)d_in[3];
  const float* bk  = (const float*)d_in[4];
  const float* wv  = (const float*)d_in[5];
  const float* bv  = (const float*)d_in[6];
  const float* wo  = (const float*)d_in[7];
  const float* bo  = (const float*)d_in[8];
  const float* rel = (const float*)d_in[9];
  const float* w1  = (const float*)d_in[10];
  const float* b1  = (const float*)d_in[11];
  const float* w2  = (const float*)d_in[12];
  const float* b2  = (const float*)d_in[13];
  const float* g1  = (const float*)d_in[14];
  const float* be1 = (const float*)d_in[15];
  const float* g2  = (const float*)d_in[16];
  const float* be2 = (const float*)d_in[17];
  float* out = (float*)d_out;

  char* p = (char*)d_ws;
  auto alloc = [&](size_t bytes) -> char* {
    char* r = p; p += (bytes + 255) & ~(size_t)255; return r;
  };
  _Float16* src_h = (_Float16*)alloc((size_t)Mrows * Dm * 2);
  _Float16* wq_h  = (_Float16*)alloc((size_t)Dm * Dm * 2);
  _Float16* wk_h  = (_Float16*)alloc((size_t)Dm * Dm * 2);
  _Float16* wv_h  = (_Float16*)alloc((size_t)Dm * Dm * 2);
  _Float16* wo_h  = (_Float16*)alloc((size_t)Dm * Dm * 2);
  _Float16* w1_h  = (_Float16*)alloc((size_t)DFFn * Dm * 2);
  _Float16* w2_h  = (_Float16*)alloc((size_t)Dm * DFFn * 2);
  _Float16* rel_h = (_Float16*)alloc((size_t)11 * 64 * 2);
  _Float16* q_h   = (_Float16*)alloc((size_t)Mrows * Dm * 2);
  _Float16* k_h   = (_Float16*)alloc((size_t)Mrows * Dm * 2);
  _Float16* v_h   = (_Float16*)alloc((size_t)Mrows * Dm * 2);
  _Float16* vt_h  = (_Float16*)alloc((size_t)Mrows * Dm * 2);
  _Float16* at_h  = (_Float16*)alloc((size_t)Mrows * Dm * 2);
  float*    z_f   = (float*)   alloc((size_t)Mrows * Dm * 4);
  float*    x_f   = (float*)   alloc((size_t)Mrows * Dm * 4);
  _Float16* x_h   = (_Float16*)alloc((size_t)Mrows * Dm * 2);
  _Float16* ff_h  = (_Float16*)alloc((size_t)Mrows * DFFn * 2);
  float*    f_f   = (float*)   alloc((size_t)Mrows * Dm * 4);

  auto cast = [&](const float* in, _Float16* o, int n) {
    cast_f2h<<<(n + 255) / 256, 256, 0, stream>>>(in, o, n);
  };
  cast(src, src_h, Mrows * Dm);
  cast(wq, wq_h, Dm * Dm);
  cast(wk, wk_h, Dm * Dm);
  cast(wv, wv_h, Dm * Dm);
  cast(wo, wo_h, Dm * Dm);
  cast(w1, w1_h, DFFn * Dm);
  cast(w2, w2_h, Dm * DFFn);
  cast(rel, rel_h, 11 * 64);

  const dim3 blk(128);
  const dim3 gProj(Dm / 128, Mrows / 128);     // (8, 32)
  gemm_kernel<false, true ><<<gProj, blk, 0, stream>>>(src_h, wq_h, bq, nullptr, q_h, Mrows, Dm, Dm);
  gemm_kernel<false, true ><<<gProj, blk, 0, stream>>>(src_h, wk_h, bk, nullptr, k_h, Mrows, Dm, Dm);
  gemm_kernel<false, true ><<<gProj, blk, 0, stream>>>(src_h, wv_h, bv, nullptr, v_h, Mrows, Dm, Dm);

  transpose_v<<<dim3(Ss / 32, DHd / 32, Bb * NH), dim3(32, 8), 0, stream>>>(v_h, vt_h);

  attn_kernel<<<dim3((Bb * NH * (Ss / 16)) / 8), dim3(256), 0, stream>>>(q_h, k_h, vt_h, rel_h, at_h);

  gemm_kernel<false, false><<<gProj, blk, 0, stream>>>(at_h, wo_h, bo, z_f, nullptr, Mrows, Dm, Dm);

  ln_kernel<<<dim3(Mrows), dim3(256), 0, stream>>>(src, z_f, g1, be1, x_f, x_h);

  gemm_kernel<true,  true ><<<dim3(DFFn / 128, Mrows / 128), blk, 0, stream>>>(x_h, w1_h, b1, nullptr, ff_h, Mrows, DFFn, Dm);
  gemm_kernel<false, false><<<gProj, blk, 0, stream>>>(ff_h, w2_h, b2, f_f, nullptr, Mrows, Dm, DFFn);

  ln_kernel<<<dim3(Mrows), dim3(256), 0, stream>>>(x_f, f_f, g2, be2, out, nullptr);
}
